// MoE_20985210208399
// MI455X (gfx1250) — matrix operand order repaired and checked by reading, not compiled, not run on hardware
//
#include <hip/hip_runtime.h>
#include <math.h>

// ---------------- problem constants (from reference) ----------------
#define T_TOKENS 16384      // B*S = 4*4096
#define H_DIM    1024
#define I_DIM    4096
#define E_NUM    8
#define TOPK     2

// ---------------- GEMM tiling ----------------
#define TM 128
#define TN 128
#define TK 32               // matches v_wmma_f32_16x16x32_bf16 K

typedef __attribute__((ext_vector_type(16))) __bf16 v16bf;
typedef __attribute__((ext_vector_type(8)))  __bf16 v8bf;
typedef __attribute__((ext_vector_type(8)))  float  v8f;

// A-matrix 16x32 bf16 fragment: lanes 0-15 -> M=lane, K 0..15 contiguous;
// lanes 16-31 -> K 16..31 (same k order as the B fragment).
__device__ __forceinline__ v16bf load_a_frag(const __bf16* rowp, int kg) {
  return *(const v16bf*)(rowp + kg);        // K = kg .. kg+15
}

// B-matrix 32x16 bf16 fragment (contiguous k order, same as A): lanes 0-15
// hold N=lane, K=0..15 contiguous; lanes 16-31 hold K=16..31.
__device__ __forceinline__ v16bf load_b_frag(const __bf16* colp, int kb) {
  return *(const v16bf*)(colp + kb);
}

__device__ __forceinline__ v8f wmma_bf16(v16bf a, v16bf b, v8f c) {
  // (neg_a, A, neg_b, B, c_mod, C, reuse_a, reuse_b)
  return __builtin_amdgcn_wmma_f32_16x16x32_bf16(false, a, false, b, (short)0, c,
                                                 false, false);
}

// gfx1250 async memory->LDS copy (CDNA5 ISA): per-lane 16B,
// LDS address in VDST VGPR (low 32 bits of the flat LDS aperture address),
// tracked by ASYNCcnt.
__device__ __forceinline__ void async_copy_b128(const void* gsrc, unsigned lds_off) {
  asm volatile("global_load_async_to_lds_b128 %0, %1, off"
               :: "v"(lds_off), "v"(gsrc)
               : "memory");
}
__device__ __forceinline__ void wait_async_all() {
  asm volatile("s_wait_asynccnt 0x0" ::: "memory");
}
__device__ __forceinline__ unsigned lds_off_of(const void* p) {
  // generic addr of LDS = {aperture_hi32, lds_byte_addr}; low 32 bits are the
  // LDS address the async instructions consume.
  return (unsigned)(size_t)p;
}

// ---------------- conversion / packing kernels ----------------

__global__ __launch_bounds__(256) void cvt_x_kernel(const float* __restrict__ in,
                                                    __bf16* __restrict__ out) {
  int i = blockIdx.x * 256 + threadIdx.x;            // i indexes float4 groups
  float4 v = ((const float4*)in)[i];
  __bf16* o = out + (size_t)i * 4;
  o[0] = (__bf16)v.x; o[1] = (__bf16)v.y; o[2] = (__bf16)v.z; o[3] = (__bf16)v.w;
}

// in: [E][R][C] fp32  ->  out: [E][C][R] bf16 (LDS-tiled transpose)
__global__ __launch_bounds__(256) void transpose_cvt_kernel(
    const float* __restrict__ in, __bf16* __restrict__ out, int R, int C) {
  __shared__ float tile[32][33];
  const int e = blockIdx.z;
  const int c0 = blockIdx.x * 32;
  const int r0 = blockIdx.y * 32;
  const float* src = in + (size_t)e * R * C;
  __bf16* dst = out + (size_t)e * R * C;
  const int tx = threadIdx.x;        // 0..31
  const int ty = threadIdx.y;        // 0..7
  #pragma unroll
  for (int i = 0; i < 32; i += 8)
    tile[ty + i][tx] = src[(size_t)(r0 + ty + i) * C + c0 + tx];
  __syncthreads();
  #pragma unroll
  for (int i = 0; i < 32; i += 8)
    dst[(size_t)(c0 + ty + i) * R + r0 + tx] = (__bf16)tile[tx][ty + i];
}

// ---------------- routing ----------------

__global__ __launch_bounds__(256) void init_kernel(int* __restrict__ counts) {
  if (threadIdx.x < E_NUM) counts[threadIdx.x] = 0;
}

// one wave32 per token: logits -> softmax -> top2 -> renorm * alpha
__global__ __launch_bounds__(256) void gate_kernel(
    const float* __restrict__ x, const float* __restrict__ gw,
    const float* __restrict__ gb, const float* __restrict__ alpha,
    int* __restrict__ top_idx, float* __restrict__ top_score,
    int* __restrict__ counts) {
  const int lane = threadIdx.x & 31;
  const int wave = threadIdx.x >> 5;
  const int t = blockIdx.x * 8 + wave;
  const float* xr = x + (size_t)t * H_DIM;
  float acc[E_NUM] = {0.f, 0.f, 0.f, 0.f, 0.f, 0.f, 0.f, 0.f};
  for (int h = lane; h < H_DIM; h += 32) {
    float xv = xr[h];
    const float* g = gw + h * E_NUM;
    #pragma unroll
    for (int e = 0; e < E_NUM; ++e) acc[e] += xv * g[e];
  }
  #pragma unroll
  for (int e = 0; e < E_NUM; ++e) {
    #pragma unroll
    for (int o = 16; o > 0; o >>= 1) acc[e] += __shfl_xor(acc[e], o, 32);
  }
  if (lane == 0) {
    float lg[E_NUM], mx = -1e30f;
    #pragma unroll
    for (int e = 0; e < E_NUM; ++e) { lg[e] = acc[e] + gb[e]; mx = fmaxf(mx, lg[e]); }
    float p[E_NUM];
    #pragma unroll
    for (int e = 0; e < E_NUM; ++e) p[e] = __expf(lg[e] - mx);
    int e0 = 0;
    #pragma unroll
    for (int e = 1; e < E_NUM; ++e) if (p[e] > p[e0]) e0 = e;
    int e1 = (e0 == 0) ? 1 : 0;
    #pragma unroll
    for (int e = 0; e < E_NUM; ++e) if (e != e0 && p[e] > p[e1]) e1 = e;
    // renormalized top-2 scores (softmax denominator cancels), scaled by alpha
    const float inv = 1.0f / (p[e0] + p[e1]);
    top_idx[2 * t]     = e0;
    top_idx[2 * t + 1] = e1;
    top_score[2 * t]     = p[e0] * inv * alpha[e0];
    top_score[2 * t + 1] = p[e1] * inv * alpha[e1];
    atomicAdd(&counts[e0], 1);
    atomicAdd(&counts[e1], 1);
  }
}

__global__ void scan_kernel(const int* __restrict__ counts,
                            int* __restrict__ offsets, int* __restrict__ fill) {
  if (threadIdx.x == 0) {
    int s = 0;
    for (int e = 0; e < E_NUM; ++e) { offsets[e] = s; fill[e] = s; s += counts[e]; }
    offsets[E_NUM] = s;
  }
}

__global__ __launch_bounds__(256) void scatter_kernel(
    const int* __restrict__ top_idx, int* __restrict__ fill,
    int* __restrict__ tok_ids, int* __restrict__ rows) {
  const int t = blockIdx.x * 256 + threadIdx.x;
  #pragma unroll
  for (int k = 0; k < TOPK; ++k) {
    int e = top_idx[2 * t + k];
    int slot = atomicAdd(&fill[e], 1);
    tok_ids[slot] = t;
    rows[2 * t + k] = slot;
  }
}

// ---------------- grouped GEMM: fc1 (+ exact gelu) ----------------
// hmid[row, I] = gelu(x[tok_ids[row]] @ w1t[e]^T + b1[e]), rows expert-sorted.
// Double-buffered LDS; all tile traffic goes through async memory->LDS copies.
// Rows >= cnt read a masked (always-valid) token row; their outputs are never
// stored, so the result is deterministic and correct.
__global__ __launch_bounds__(256) void fc1_kernel(
    const __bf16* __restrict__ xb, const __bf16* __restrict__ w1t,
    const float* __restrict__ b1, const int* __restrict__ tok_ids,
    const int* __restrict__ counts, const int* __restrict__ offsets,
    __bf16* __restrict__ hmid) {
  const int e = blockIdx.z;
  const int cnt = counts[e];
  const int m0 = blockIdx.x * TM;
  if (m0 >= cnt) return;                               // empty tile -> exit
  const int off = offsets[e];
  const int n0 = blockIdx.y * TN;

  __shared__ __attribute__((aligned(32))) __bf16 sA[2][TM * TK];
  __shared__ __attribute__((aligned(32))) __bf16 sB[2][TN * TK];

  const int tid  = threadIdx.x;
  const int lane = tid & 31;
  const int wave = tid >> 5;          // 8 waves
  const int wm = (wave & 3) * 32;     // wave M origin
  const int wn = (wave >> 2) * 64;    // wave N origin
  const int rl = lane & 15;
  const int kg = (lane >> 4) * 16;    // A frag K group
  const int kb = (lane >> 4) * 16;    // B frag K group

  // ---- hoisted per-thread tile-staging addresses (2 rows x 16B each) ----
  const int r0 = tid >> 2;            // 0..63
  const int r1 = r0 + 64;             // 64..127
  const int kc = (tid & 3) * 8;       // K chunk within tile
  // tok_ids is padded; mask keeps gathered row in-bounds for rows >= cnt.
  const int t0 = tok_ids[off + m0 + r0] & (T_TOKENS - 1);
  const int t1 = tok_ids[off + m0 + r1] & (T_TOKENS - 1);
  const __bf16* ag0 = xb + (size_t)t0 * H_DIM + kc;
  const __bf16* ag1 = xb + (size_t)t1 * H_DIM + kc;
  const __bf16* bg0 = w1t + ((size_t)e * I_DIM + n0 + r0) * H_DIM + kc;
  const __bf16* bg1 = w1t + ((size_t)e * I_DIM + n0 + r1) * H_DIM + kc;
  unsigned la0[2], la1[2], lb0[2], lb1[2];
  #pragma unroll
  for (int b = 0; b < 2; ++b) {
    la0[b] = lds_off_of(&sA[b][r0 * TK + kc]);
    la1[b] = lds_off_of(&sA[b][r1 * TK + kc]);
    lb0[b] = lds_off_of(&sB[b][r0 * TK + kc]);
    lb1[b] = lds_off_of(&sB[b][r1 * TK + kc]);
  }

  v8f acc[2][4];
  const v8f vzero = {};
  #pragma unroll
  for (int mi = 0; mi < 2; ++mi)
    #pragma unroll
    for (int ni = 0; ni < 4; ++ni) acc[mi][ni] = vzero;

  // prologue: stage tile 0 into buffer 0
  async_copy_b128(ag0, la0[0]);
  async_copy_b128(ag1, la1[0]);
  async_copy_b128(bg0, lb0[0]);
  async_copy_b128(bg1, lb1[0]);
  wait_async_all();
  __syncthreads();

  int buf = 0;
  for (int k0 = 0; k0 < H_DIM; k0 += TK) {
    const bool nxt = (k0 + TK) < H_DIM;
    if (nxt) {   // issue next tile into the other buffer (overlaps WMMAs)
      async_copy_b128(ag0 + k0 + TK, la0[buf ^ 1]);
      async_copy_b128(ag1 + k0 + TK, la1[buf ^ 1]);
      async_copy_b128(bg0 + k0 + TK, lb0[buf ^ 1]);
      async_copy_b128(bg1 + k0 + TK, lb1[buf ^ 1]);
    }
    v16bf afr[2], bfr[4];
    #pragma unroll
    for (int mi = 0; mi < 2; ++mi)
      afr[mi] = load_a_frag(&sA[buf][(wm + mi * 16 + rl) * TK], kg);
    #pragma unroll
    for (int ni = 0; ni < 4; ++ni)
      bfr[ni] = load_b_frag(&sB[buf][(wn + ni * 16 + rl) * TK], kb);
    #pragma unroll
    for (int mi = 0; mi < 2; ++mi)
      #pragma unroll
      for (int ni = 0; ni < 4; ++ni)
        acc[mi][ni] = wmma_bf16(afr[mi], bfr[ni], acc[mi][ni]);
    if (nxt) wait_async_all();
    __syncthreads();
    buf ^= 1;
  }

  // epilogue: C/D layout -> VGPR j holds M=j (lanes 0-15) / M=j+8 (lanes 16-31)
  const int mhi = (lane >> 4) * 8;
  #pragma unroll
  for (int mi = 0; mi < 2; ++mi) {
    #pragma unroll
    for (int ni = 0; ni < 4; ++ni) {
      int ncol = n0 + wn + ni * 16 + rl;
      float bias = b1[e * I_DIM + ncol];
      #pragma unroll
      for (int j = 0; j < 8; ++j) {
        int m = wm + mi * 16 + mhi + j;
        if (m0 + m < cnt) {
          float v = acc[mi][ni][j] + bias;
          v = 0.5f * v * (1.0f + erff(v * 0.70710678118654752f));   // exact gelu
          hmid[(size_t)(off + m0 + m) * I_DIM + ncol] = (__bf16)v;
        }
      }
    }
  }
}

// ---------------- grouped GEMM: fc2 ----------------
// ybuf[row, H] = hmid[row] @ w2t[e]^T  (bias/score applied in combine)
__global__ __launch_bounds__(256) void fc2_kernel(
    const __bf16* __restrict__ hmid, const __bf16* __restrict__ w2t,
    const int* __restrict__ counts, const int* __restrict__ offsets,
    float* __restrict__ ybuf) {
  const int e = blockIdx.z;
  const int cnt = counts[e];
  const int m0 = blockIdx.x * TM;
  if (m0 >= cnt) return;
  const int off = offsets[e];
  const int n0 = blockIdx.y * TN;

  __shared__ __attribute__((aligned(32))) __bf16 sA[2][TM * TK];
  __shared__ __attribute__((aligned(32))) __bf16 sB[2][TN * TK];

  const int tid  = threadIdx.x;
  const int lane = tid & 31;
  const int wave = tid >> 5;
  const int wm = (wave & 3) * 32;
  const int wn = (wave >> 2) * 64;
  const int rl = lane & 15;
  const int kg = (lane >> 4) * 16;
  const int kb = (lane >> 4) * 16;

  const int r0 = tid >> 2;
  const int r1 = r0 + 64;
  const int kc = (tid & 3) * 8;
  // hmid is padded by >=TM rows, so rows >= cnt are safe (discarded) reads.
  const __bf16* ag0 = hmid + (size_t)(off + m0 + r0) * I_DIM + kc;
  const __bf16* ag1 = hmid + (size_t)(off + m0 + r1) * I_DIM + kc;
  const __bf16* bg0 = w2t + ((size_t)e * H_DIM + n0 + r0) * I_DIM + kc;
  const __bf16* bg1 = w2t + ((size_t)e * H_DIM + n0 + r1) * I_DIM + kc;
  unsigned la0[2], la1[2], lb0[2], lb1[2];
  #pragma unroll
  for (int b = 0; b < 2; ++b) {
    la0[b] = lds_off_of(&sA[b][r0 * TK + kc]);
    la1[b] = lds_off_of(&sA[b][r1 * TK + kc]);
    lb0[b] = lds_off_of(&sB[b][r0 * TK + kc]);
    lb1[b] = lds_off_of(&sB[b][r1 * TK + kc]);
  }

  v8f acc[2][4];
  const v8f vzero = {};
  #pragma unroll
  for (int mi = 0; mi < 2; ++mi)
    #pragma unroll
    for (int ni = 0; ni < 4; ++ni) acc[mi][ni] = vzero;

  async_copy_b128(ag0, la0[0]);
  async_copy_b128(ag1, la1[0]);
  async_copy_b128(bg0, lb0[0]);
  async_copy_b128(bg1, lb1[0]);
  wait_async_all();
  __syncthreads();

  int buf = 0;
  for (int k0 = 0; k0 < I_DIM; k0 += TK) {
    const bool nxt = (k0 + TK) < I_DIM;
    if (nxt) {
      async_copy_b128(ag0 + k0 + TK, la0[buf ^ 1]);
      async_copy_b128(ag1 + k0 + TK, la1[buf ^ 1]);
      async_copy_b128(bg0 + k0 + TK, lb0[buf ^ 1]);
      async_copy_b128(bg1 + k0 + TK, lb1[buf ^ 1]);
    }
    v16bf afr[2], bfr[4];
    #pragma unroll
    for (int mi = 0; mi < 2; ++mi)
      afr[mi] = load_a_frag(&sA[buf][(wm + mi * 16 + rl) * TK], kg);
    #pragma unroll
    for (int ni = 0; ni < 4; ++ni)
      bfr[ni] = load_b_frag(&sB[buf][(wn + ni * 16 + rl) * TK], kb);
    #pragma unroll
    for (int mi = 0; mi < 2; ++mi)
      #pragma unroll
      for (int ni = 0; ni < 4; ++ni)
        acc[mi][ni] = wmma_bf16(afr[mi], bfr[ni], acc[mi][ni]);
    if (nxt) wait_async_all();
    __syncthreads();
    buf ^= 1;
  }

  const int mhi = (lane >> 4) * 8;
  #pragma unroll
  for (int mi = 0; mi < 2; ++mi) {
    #pragma unroll
    for (int ni = 0; ni < 4; ++ni) {
      int ncol = n0 + wn + ni * 16 + rl;
      #pragma unroll
      for (int j = 0; j < 8; ++j) {
        int m = wm + mi * 16 + mhi + j;
        if (m0 + m < cnt)
          ybuf[(size_t)(off + m0 + m) * H_DIM + ncol] = acc[mi][ni][j];
      }
    }
  }
}

// ---------------- combine: out[t] = s0*(y0+b2[e0]) + s1*(y1+b2[e1]) ----------------
__global__ __launch_bounds__(256) void combine_kernel(
    const float* __restrict__ ybuf, const float* __restrict__ b2,
    const int* __restrict__ top_idx, const float* __restrict__ top_score,
    const int* __restrict__ rows, float* __restrict__ out) {
  size_t i = (size_t)blockIdx.x * 256 + threadIdx.x;
  int t = (int)(i >> 10);                 // H_DIM == 1024
  int h = (int)(i & (H_DIM - 1));
  int r0 = rows[2 * t], r1 = rows[2 * t + 1];
  int e0 = top_idx[2 * t], e1 = top_idx[2 * t + 1];
  float y0 = ybuf[(size_t)r0 * H_DIM + h] + b2[e0 * H_DIM + h];
  float y1 = ybuf[(size_t)r1 * H_DIM + h] + b2[e1 * H_DIM + h];
  out[i] = top_score[2 * t] * y0 + top_score[2 * t + 1] * y1;
}

// ---------------- launch ----------------
extern "C" void kernel_launch(void* const* d_in, const int* in_sizes, int n_in,
                              void* d_out, int out_size, void* d_ws, size_t ws_size,
                              hipStream_t stream) {
  (void)in_sizes; (void)n_in; (void)out_size; (void)ws_size;
  const float* x      = (const float*)d_in[0];   // [B,S,H]
  const float* gate_w = (const float*)d_in[1];   // [H,E]
  const float* gate_b = (const float*)d_in[2];   // [E]
  const float* w1     = (const float*)d_in[3];   // [E,H,I]
  const float* b1     = (const float*)d_in[4];   // [E,I]
  const float* w2     = (const float*)d_in[5];   // [E,I,H]
  const float* b2     = (const float*)d_in[6];   // [E,H]
  const float* alpha  = (const float*)d_in[7];   // [E]
  float* out = (float*)d_out;

  char* ws = (char*)d_ws;
  size_t o = 0;
  auto take = [&](size_t bytes) -> void* {
    void* p = ws + o;
    o += (bytes + 255) & ~(size_t)255;
    return p;
  };
  __bf16* xb   = (__bf16*)take((size_t)T_TOKENS * H_DIM * 2);
  __bf16* w1t  = (__bf16*)take((size_t)E_NUM * I_DIM * H_DIM * 2);  // [E][I][H]
  __bf16* w2t  = (__bf16*)take((size_t)E_NUM * H_DIM * I_DIM * 2);  // [E][H][I]
  // +TM rows of padding: tail tiles issue (discarded) in-bounds reads
  __bf16* hmid = (__bf16*)take(((size_t)TOPK * T_TOKENS + TM) * I_DIM * 2);
  float*  ybuf = (float*)take((size_t)TOPK * T_TOKENS * H_DIM * 4);
  int*   top_idx   = (int*)take((size_t)TOPK * T_TOKENS * 4);
  float* top_score = (float*)take((size_t)TOPK * T_TOKENS * 4);
  int*   rows      = (int*)take((size_t)TOPK * T_TOKENS * 4);
  int*   tok_ids   = (int*)take(((size_t)TOPK * T_TOKENS + TM) * 4);  // padded
  int*   counts    = (int*)take(E_NUM * 4);
  int*   offsets   = (int*)take((E_NUM + 1) * 4);
  int*   fill      = (int*)take(E_NUM * 4);

  init_kernel<<<1, 32, 0, stream>>>(counts);
  cvt_x_kernel<<<(T_TOKENS * H_DIM) / (256 * 4), 256, 0, stream>>>(x, xb);
  {
    dim3 tb(32, 8);
    transpose_cvt_kernel<<<dim3(I_DIM / 32, H_DIM / 32, E_NUM), tb, 0, stream>>>(
        w1, w1t, H_DIM, I_DIM);
    transpose_cvt_kernel<<<dim3(H_DIM / 32, I_DIM / 32, E_NUM), tb, 0, stream>>>(
        w2, w2t, I_DIM, H_DIM);
  }
  gate_kernel<<<T_TOKENS / 8, 256, 0, stream>>>(x, gate_w, gate_b, alpha,
                                                top_idx, top_score, counts);
  scan_kernel<<<1, 32, 0, stream>>>(counts, offsets, fill);
  scatter_kernel<<<T_TOKENS / 256, 256, 0, stream>>>(top_idx, fill, tok_ids, rows);

  fc1_kernel<<<dim3(T_TOKENS / TM, I_DIM / TN, E_NUM), 256, 0, stream>>>(
      xb, w1t, b1, tok_ids, counts, offsets, hmid);
  fc2_kernel<<<dim3(T_TOKENS / TM, H_DIM / TN, E_NUM), 256, 0, stream>>>(
      hmid, w2t, counts, offsets, ybuf);

  combine_kernel<<<(T_TOKENS * H_DIM) / 256, 256, 0, stream>>>(
      ybuf, b2, top_idx, top_score, rows, out);
}
